// MoE_23983097381213
// MI455X (gfx1250) — matrix operand order repaired and checked by reading, not compiled, not run on hardware
//
#include <hip/hip_runtime.h>

#define N_TOK 4096
#define DIM   1024
#define HID   4096
#define NEXP  8
#define KPAD  40   // row stride 80B: 16B-aligned fragments, spreads LDS banks

typedef __attribute__((ext_vector_type(16))) __bf16 v16bf;
typedef __attribute__((ext_vector_type(8)))  float  v8f;

typedef __attribute__((address_space(1))) int gint_t;
typedef __attribute__((address_space(3))) int lint_t;

union FragU { v16bf v; uint4 q[2]; };

// One-instruction packed f32->bf16 (RNE), mnemonic verified from compiler output.
__device__ __forceinline__ unsigned pk2(float a, float b) {
  unsigned u;
  asm("v_cvt_pk_bf16_f32 %0, %1, %2" : "=v"(u) : "v"(a), "v"(b));
  return u;
}

// ---------------------------------------------------------------------------
// Kernel 1: router. One wave (32 lanes) per token. w_gate cached transposed
// in LDS. Emits per-token (expert0, expert1, gate0, gate1).
// ---------------------------------------------------------------------------
__global__ __launch_bounds__(256) void moe_route(
    const float* __restrict__ x, const float* __restrict__ w_gate,
    int* __restrict__ tokE, float* __restrict__ tokG)
{
  __shared__ float wg[NEXP * DIM];   // [e][d]
  const int tid = threadIdx.x;
  for (int i = tid; i < NEXP * DIM; i += 256) {
    const int d = i >> 3, e = i & 7;
    wg[e * DIM + d] = w_gate[i];
  }
  __syncthreads();

  const int lane  = tid & 31;
  const int token = blockIdx.x * 8 + (tid >> 5);
  const float* xr = x + (size_t)token * DIM;

  float acc[NEXP];
#pragma unroll
  for (int e = 0; e < NEXP; ++e) acc[e] = 0.f;

  for (int d = lane; d < DIM; d += 32) {
    const float xv = xr[d];
#pragma unroll
    for (int e = 0; e < NEXP; ++e) acc[e] += xv * wg[e * DIM + d];
  }
#pragma unroll
  for (int off = 16; off > 0; off >>= 1) {
#pragma unroll
    for (int e = 0; e < NEXP; ++e) acc[e] += __shfl_xor(acc[e], off, 32);
  }

  if (lane == 0) {
    int i0 = 0;
#pragma unroll
    for (int e = 1; e < NEXP; ++e) if (acc[e] > acc[i0]) i0 = e;
    int i1 = (i0 == 0) ? 1 : 0;
#pragma unroll
    for (int e = 0; e < NEXP; ++e) if (e != i0 && acc[e] > acc[i1]) i1 = e;
    const float g0 = 1.f / (1.f + __expf(acc[i1] - acc[i0]));  // softmax top-2
    tokE[2 * token]     = i0;
    tokE[2 * token + 1] = i1;
    tokG[2 * token]     = g0;
    tokG[2 * token + 1] = 1.f - g0;
  }
}

// ---------------------------------------------------------------------------
// Kernel 2: deterministic per-expert compaction (8 waves = 8 experts) +
// importance/load stats + CV^2 aux loss.
// ---------------------------------------------------------------------------
__global__ __launch_bounds__(256) void moe_build(
    const int* __restrict__ tokE, const float* __restrict__ tokG,
    int* __restrict__ lists, float* __restrict__ listg,
    int* __restrict__ cnt, float* __restrict__ lossOut)
{
  const int lane = threadIdx.x & 31;
  const int e    = threadIdx.x >> 5;

  __shared__ float impS[NEXP];
  __shared__ float loadS[NEXP];

  int   base   = 0;
  float impSum = 0.f;

  for (int t0 = 0; t0 < N_TOK; t0 += 32) {
    const int t  = t0 + lane;
    const int e0 = tokE[2 * t];
    const int e1 = tokE[2 * t + 1];
    const float g0 = tokG[2 * t];
    const float g1 = tokG[2 * t + 1];

    const bool m0 = (e0 == e);
    const unsigned mask0 = (unsigned)__ballot(m0);
    if (m0) {
      const int pos = base + __popc(mask0 & ((1u << lane) - 1u));
      lists[e * N_TOK + pos] = t;
      listg[e * N_TOK + pos] = g0;
      impSum += g0;
    }
    base += __popc(mask0);

    const bool m1 = (e1 == e);
    const unsigned mask1 = (unsigned)__ballot(m1);
    if (m1) {
      const int pos = base + __popc(mask1 & ((1u << lane) - 1u));
      lists[e * N_TOK + pos] = t;
      listg[e * N_TOK + pos] = g1;
      impSum += g1;
    }
    base += __popc(mask1);
  }

#pragma unroll
  for (int off = 16; off > 0; off >>= 1) impSum += __shfl_xor(impSum, off, 32);

  if (lane == 0) {
    cnt[e]   = base;
    impS[e]  = impSum;
    loadS[e] = (float)base;
  }
  __syncthreads();

  if (threadIdx.x == 0) {
    float mi = 0.f, ml = 0.f;
#pragma unroll
    for (int k = 0; k < NEXP; ++k) { mi += impS[k]; ml += loadS[k]; }
    mi *= (1.f / NEXP); ml *= (1.f / NEXP);
    float vi = 0.f, vl = 0.f;
#pragma unroll
    for (int k = 0; k < NEXP; ++k) {
      float d = impS[k] - mi;  vi += d * d;
      d = loadS[k] - ml;       vl += d * d;
    }
    vi *= (1.f / (NEXP - 1)); vl *= (1.f / (NEXP - 1));   // ddof=1
    lossOut[0] = 0.01f * (vi / (mi * mi + 1e-10f) + vl / (ml * ml + 1e-10f));
  }
}

// ---------------------------------------------------------------------------
// Kernel 3: gathered expert GEMM, bf16 WMMA + f32 accumulate.
// 256 threads (8 waves, 2x4), 128x128 tile, K-step 32,
// double-buffered LDS with register prefetch (1 barrier / K-step).
// ---------------------------------------------------------------------------
__global__ __launch_bounds__(256) void moe_gemm(
    const float* __restrict__ x, const float* __restrict__ expert_w,
    const float* __restrict__ expert_b,
    const int* __restrict__ lists, const float* __restrict__ listg,
    const int* __restrict__ cnt, float* __restrict__ y)
{
  const int e  = blockIdx.z;
  const int n0 = blockIdx.x * 128;
  const int m0 = blockIdx.y * 128;

  const int count = cnt[e];
  if (m0 >= count) return;                       // block-uniform exit
  const int rows = min(128, count - m0);

  __shared__ __bf16 As[2][128][KPAD];            // [m][k]
  __shared__ __bf16 Bs[2][128][KPAD];            // [n][k]
  __shared__ int   tokS[128];
  __shared__ float gateS[128];

  const int tid   = threadIdx.x;
  const int lane  = tid & 31;
  const int wave  = tid >> 5;
  const int waveM = wave & 1;                    // 2 waves along M
  const int waveN = wave >> 1;                   // 4 waves along N

  // ---- stage token ids / gates via gfx1250 async global->LDS copies ----
#if __has_builtin(__builtin_amdgcn_global_load_async_to_lds_b32) && \
    __has_builtin(__builtin_amdgcn_s_wait_asynccnt)
  if (tid < 128) {
    __builtin_amdgcn_global_load_async_to_lds_b32(
        (gint_t*)(lists + (size_t)e * N_TOK + m0 + tid),
        (lint_t*)&tokS[tid], 0, 0);
    __builtin_amdgcn_global_load_async_to_lds_b32(
        (gint_t*)(listg + (size_t)e * N_TOK + m0 + tid),
        (lint_t*)&gateS[tid], 0, 0);
  }
  __builtin_amdgcn_s_wait_asynccnt(0);
#else
  if (tid < 128) {
    tokS[tid]  = lists[(size_t)e * N_TOK + m0 + tid];   // entries >= rows unused
    gateS[tid] = listg[(size_t)e * N_TOK + m0 + tid];
  }
#endif
  __syncthreads();

  // Accumulators initialized with expert bias (y += g*(xW + b)).
  v8f acc[4][2];
#pragma unroll
  for (int j = 0; j < 2; ++j) {
    const int ng = n0 + waveN * 32 + j * 16 + (lane & 15);
    const float b = expert_b[e * HID + ng];
#pragma unroll
    for (int i = 0; i < 4; ++i) {
#pragma unroll
      for (int r = 0; r < 8; ++r) acc[i][j][r] = b;
    }
  }

  // A loader: 2 threads per tile row, 16 cols each.
  const int ar = tid >> 1;
  const int ac = (tid & 1) << 4;
  const bool aValid = (ar < rows);
  const float* xrow = x + (size_t)(aValid ? tokS[ar] : 0) * DIM;

  // B loader: wave owns 4 K-rows, lane owns 4 N-cols; transpose in registers.
  const float* wrow = expert_w + ((size_t)e * DIM + wave * 4) * HID + n0 + lane * 4;

  float4 fa[4], fb[4];

  auto gloadA = [&](int kb) {
    if (aValid) {
      const float4* xp = (const float4*)(xrow + kb * 32 + ac);
      fa[0] = xp[0]; fa[1] = xp[1]; fa[2] = xp[2]; fa[3] = xp[3];
    } else {
      fa[0] = fa[1] = fa[2] = fa[3] = make_float4(0.f, 0.f, 0.f, 0.f);
    }
  };
  auto gloadB = [&](int kb) {
#pragma unroll
    for (int r = 0; r < 4; ++r)
      fb[r] = *(const float4*)(wrow + (size_t)(kb * 32 + r) * HID);
  };
  auto storeTiles = [&](int buf) {
    uint4 p0, p1;
    p0.x = pk2(fa[0].x, fa[0].y); p0.y = pk2(fa[0].z, fa[0].w);
    p0.z = pk2(fa[1].x, fa[1].y); p0.w = pk2(fa[1].z, fa[1].w);
    p1.x = pk2(fa[2].x, fa[2].y); p1.y = pk2(fa[2].z, fa[2].w);
    p1.z = pk2(fa[3].x, fa[3].y); p1.w = pk2(fa[3].z, fa[3].w);
    *(uint4*)&As[buf][ar][ac]     = p0;
    *(uint4*)&As[buf][ar][ac + 8] = p1;
#pragma unroll
    for (int j = 0; j < 4; ++j) {                 // pack along K -> b64 stores
      uint2 p;
      p.x = pk2(((const float*)&fb[0])[j], ((const float*)&fb[1])[j]);
      p.y = pk2(((const float*)&fb[2])[j], ((const float*)&fb[3])[j]);
      *(uint2*)&Bs[buf][lane * 4 + j][wave * 4] = p;
    }
  };

  // ---- prologue: fill buffer 0 ----
  gloadA(0); gloadB(0);
  storeTiles(0);
  __syncthreads();

  for (int kb = 0; kb < DIM / 32; ++kb) {
    const int  cur  = kb & 1;
    const bool more = (kb + 1) < (DIM / 32);

    if (more) { gloadA(kb + 1); gloadB(kb + 1); }     // overlap with compute
    if (kb + 2 < DIM / 32) {                           // L2 prefetch 2 steps out
      __builtin_prefetch(wrow + (size_t)((kb + 2) * 32) * HID, 0, 1);
      if (aValid) __builtin_prefetch(xrow + (kb + 2) * 32 + ac, 0, 1);
    }

    // ---- fragments then 8 WMMAs ----
    FragU bfrag[2], afrag[4];
#pragma unroll
    for (int j = 0; j < 2; ++j) {
      const int n  = waveN * 32 + j * 16 + (lane & 15);
      const int kB = (lane >> 4) << 3;               // chunks at kB, kB+16
      bfrag[j].q[0] = *(const uint4*)&Bs[cur][n][kB];
      bfrag[j].q[1] = *(const uint4*)&Bs[cur][n][kB + 16];
    }
#pragma unroll
    for (int i = 0; i < 4; ++i) {
      const int m  = waveM * 64 + i * 16 + (lane & 15);
      const int k0 = (lane >> 4) << 3;               // chunks at k0, k0+16
      afrag[i].q[0] = *(const uint4*)&As[cur][m][k0];
      afrag[i].q[1] = *(const uint4*)&As[cur][m][k0 + 16];
    }
#pragma unroll
    for (int i = 0; i < 4; ++i) {
#pragma unroll
      for (int j = 0; j < 2; ++j) {
        acc[i][j] = __builtin_amdgcn_wmma_f32_16x16x32_bf16(
            false, afrag[i].v, false, bfrag[j].v, (short)0, acc[i][j],
            false, false);
      }
    }

    if (more) storeTiles(cur ^ 1);                   // fill other buffer
    __syncthreads();
  }

  // ---- epilogue: gate-scale + scatter-accumulate into y ----
#pragma unroll
  for (int i = 0; i < 4; ++i) {
    const int mSub = waveM * 64 + i * 16 + ((lane >> 4) << 3);
#pragma unroll
    for (int j = 0; j < 2; ++j) {
      const int ng = n0 + waveN * 32 + j * 16 + (lane & 15);
#pragma unroll
      for (int r = 0; r < 8; ++r) {
        const int m = mSub + r;
        if (m < rows) {
          atomicAdd(&y[(size_t)tokS[m] * HID + ng], gateS[m] * acc[i][j][r]);
        }
      }
    }
  }
}

// ---------------------------------------------------------------------------
extern "C" void kernel_launch(void* const* d_in, const int* in_sizes, int n_in,
                              void* d_out, int out_size, void* d_ws, size_t ws_size,
                              hipStream_t stream)
{
  (void)in_sizes; (void)n_in; (void)ws_size;

  const float* x        = (const float*)d_in[0];
  const float* w_gate   = (const float*)d_in[1];
  const float* expert_w = (const float*)d_in[2];
  const float* expert_b = (const float*)d_in[3];

  float* y    = (float*)d_out;
  float* loss = y + (size_t)N_TOK * HID;

  char* ws = (char*)d_ws;
  int*   tokE  = (int*)ws;    ws += (size_t)2 * N_TOK * sizeof(int);
  float* tokG  = (float*)ws;  ws += (size_t)2 * N_TOK * sizeof(float);
  int*   lists = (int*)ws;    ws += (size_t)NEXP * N_TOK * sizeof(int);
  float* listg = (float*)ws;  ws += (size_t)NEXP * N_TOK * sizeof(float);
  int*   cnt   = (int*)ws;

  (void)hipMemsetAsync(d_out, 0, (size_t)out_size * sizeof(float), stream);

  moe_route<<<N_TOK / 8, 256, 0, stream>>>(x, w_gate, tokE, tokG);
  moe_build<<<1, 256, 0, stream>>>(tokE, tokG, lists, listg, cnt, loss);
  moe_gemm<<<dim3(HID / 128, N_TOK / 128, NEXP), 256, 0, stream>>>(
      x, expert_w, expert_b, lists, listg, cnt, y);
}
